// AttentionBlock_14929306320970
// MI455X (gfx1250) — matrix operand order repaired and checked by reading, not compiled, not run on hardware
//
#include <hip/hip_runtime.h>
#include <hip/hip_bf16.h>
#include <math.h>

// ---------------- CDNA5 WMMA types ----------------
typedef __attribute__((ext_vector_type(16))) __bf16 v16bf;
typedef __attribute__((ext_vector_type(8)))  float  v8f;

#define B_    16
#define C_    256
#define N_    1024        // 32*32 spatial
#define HEADS 2
#define HD    128
#define GROUPS 4
#define CPG   64          // channels per group

// round-to-nearest-even f32 -> bf16 (bit trick; inputs are finite)
__device__ __forceinline__ unsigned short f2bf(float f){
  unsigned u = __float_as_uint(f);
  unsigned r = u + 0x7FFFu + ((u >> 16) & 1u);
  return (unsigned short)(r >> 16);
}
__device__ __forceinline__ unsigned pk2(float a, float b){
  return (unsigned)f2bf(a) | ((unsigned)f2bf(b) << 16);
}

union ABreg { v16bf v; uint4 q[2]; unsigned u[8]; unsigned short s[16]; };

// A tile 16(M) x 32(K), bf16, memory row-major [m][k] (k contiguous), ld elems.
// lane<16 holds row m=lane, K=0..15; lane>=16: K=16..31.
__device__ __forceinline__ v16bf load_A(const unsigned short* base, int ld){
  int lane = threadIdx.x & 31;
  const unsigned short* p = base + (size_t)(lane & 15) * ld + ((lane >> 4) << 4);
  ABreg r;
  r.q[0] = *(const uint4*)p;
  r.q[1] = *(const uint4*)(p + 8);
  return r.v;
}
// B tile 32(K) x 16(N), bf16, column n stored K-contiguous ([n][k] rows), ld elems.
// lane<16 holds col n=lane, K=0..15; lane>=16 holds K=16..31.
__device__ __forceinline__ v16bf load_B(const unsigned short* base, int ld){
  int lane = threadIdx.x & 31;
  const unsigned short* p = base + (size_t)(lane & 15) * ld + ((lane >> 4) << 4);
  ABreg r;
  r.q[0] = *(const uint4*)p;
  r.q[1] = *(const uint4*)(p + 8);
  return r.v;
}
__device__ __forceinline__ v8f wmma_bf16(v16bf a, v16bf b, v8f c){
  return __builtin_amdgcn_wmma_f32_16x16x32_bf16(false, a, false, b, (short)0, c, false, false);
}

// ---------------- GroupNorm + transpose to bf16 [b][n][c] ----------------
__global__ void gn_kernel(const float* __restrict__ x, const float* __restrict__ gw,
                          const float* __restrict__ gb, unsigned short* __restrict__ xnT){
  int b = blockIdx.x >> 2, g = blockIdx.x & 3;
  int c0 = g * CPG;
  const float* xb = x + (size_t)b * C_ * N_;
  int tid = threadIdx.x;
  __shared__ float red[256];
  __shared__ float scl[CPG], bia[CPG];
  __shared__ unsigned short tile[16][CPG];

  // pass 1: mean / var over 64 channels x 1024 spatial
  float s = 0.f, s2 = 0.f;
  for (int i = 0; i < (CPG * N_) / 256; ++i){
    int e = tid + i * 256;
    float v = xb[(size_t)(c0 + (e >> 10)) * N_ + (e & (N_ - 1))];
    s += v; s2 += v * v;
  }
  red[tid] = s; __syncthreads();
  for (int st = 128; st > 0; st >>= 1){ if (tid < st) red[tid] += red[tid + st]; __syncthreads(); }
  float mean = red[0] * (1.0f / (CPG * N_));
  __syncthreads();
  red[tid] = s2; __syncthreads();
  for (int st = 128; st > 0; st >>= 1){ if (tid < st) red[tid] += red[tid + st]; __syncthreads(); }
  float var = red[0] * (1.0f / (CPG * N_)) - mean * mean;
  float inv = rsqrtf(var + 1e-5f);
  if (tid < CPG){
    float wsc = gw[c0 + tid] * inv;
    scl[tid] = wsc;
    bia[tid] = gb[c0 + tid] - mean * wsc;
  }
  __syncthreads();

  // pass 2: normalize + LDS-tiled transpose, write bf16 xnT[b][n][c]
  int cl = tid >> 2, ng = tid & 3;   // read mapping: 64c x 4 threads x float4
  int nl = tid >> 4, cg = tid & 15;  // write mapping: 16n x 16 threads x 4 halves
  for (int n0 = 0; n0 < N_; n0 += 16){
    float4 v = *(const float4*)(xb + (size_t)(c0 + cl) * N_ + n0 + ng * 4);
    float a = scl[cl], bb = bia[cl];
    int rr = ng * 4;
    tile[rr + 0][cl] = f2bf(v.x * a + bb);
    tile[rr + 1][cl] = f2bf(v.y * a + bb);
    tile[rr + 2][cl] = f2bf(v.z * a + bb);
    tile[rr + 3][cl] = f2bf(v.w * a + bb);
    __syncthreads();
    uint2 w2 = *(const uint2*)&tile[nl][cg * 4];
    *(uint2*)(xnT + (size_t)b * N_ * C_ + (size_t)(n0 + nl) * C_ + c0 + cg * 4) = w2;
    __syncthreads();
  }
}

// ---------------- weight f32 -> bf16 ----------------
__global__ void cvt_kernel(const float* __restrict__ w, unsigned short* __restrict__ o, int n){
  int i = blockIdx.x * blockDim.x + threadIdx.x;
  if (i < n) o[i] = f2bf(w[i]);
}

// ---------------- y = W @ xn, stored transposed bf16 [b][n][o] (for Q, K) ----------------
__global__ void proj_qk_kernel(const unsigned short* __restrict__ wbf,
                               const unsigned short* __restrict__ xnT,
                               unsigned short* __restrict__ outT){
  int wid = blockIdx.x * (blockDim.x >> 5) + (threadIdx.x >> 5);
  int b = wid >> 8;
  int rem = wid & 255;
  int o0 = (rem >> 4) << 4;     // 16 output channels
  int n0 = (rem & 15) << 6;     // 64 spatial cols
  const unsigned short* xb = xnT + (size_t)b * N_ * C_;
  v8f acc[4] = {};
  for (int cc = 0; cc < C_; cc += 32){
    v16bf a = load_A(wbf + (size_t)o0 * C_ + cc, C_);
    #pragma unroll
    for (int t = 0; t < 4; ++t)
      acc[t] = wmma_bf16(a, load_B(xb + (size_t)(n0 + t * 16) * C_ + cc, C_), acc[t]);
  }
  int lane = threadIdx.x & 31;
  unsigned short* ob = outT + (size_t)b * N_ * C_ + o0 + ((lane >> 4) << 3);
  #pragma unroll
  for (int t = 0; t < 4; ++t){
    int n = n0 + t * 16 + (lane & 15);
    uint4 pkd = { pk2(acc[t][0], acc[t][1]), pk2(acc[t][2], acc[t][3]),
                  pk2(acc[t][4], acc[t][5]), pk2(acc[t][6], acc[t][7]) };
    *(uint4*)(ob + (size_t)n * C_) = pkd;
  }
}

// ---------------- v = Wv @ xn, stored natural bf16 [b][o][n] (n contiguous) ----------------
// computed as D = xnT x Wv^T so D cols are output channels, rows are spatial
__global__ void proj_v_kernel(const unsigned short* __restrict__ xnT,
                              const unsigned short* __restrict__ wvbf,
                              unsigned short* __restrict__ vout){
  int wid = blockIdx.x * (blockDim.x >> 5) + (threadIdx.x >> 5);
  int b = wid >> 8;
  int rem = wid & 255;
  int n0 = (rem >> 2) << 4;     // 16 spatial rows
  int o0 = (rem & 3) << 6;      // 64 output channels
  const unsigned short* xb = xnT + (size_t)b * N_ * C_;
  v8f acc[4] = {};
  for (int cc = 0; cc < C_; cc += 32){
    v16bf a = load_A(xb + (size_t)n0 * C_ + cc, C_);
    #pragma unroll
    for (int t = 0; t < 4; ++t)
      acc[t] = wmma_bf16(a, load_B(wvbf + (size_t)(o0 + t * 16) * C_ + cc, C_), acc[t]);
  }
  int lane = threadIdx.x & 31;
  int nrow = n0 + ((lane >> 4) << 3);
  unsigned short* ob = vout + (size_t)b * C_ * N_ + nrow;
  #pragma unroll
  for (int t = 0; t < 4; ++t){
    int o = o0 + t * 16 + (lane & 15);
    uint4 pkd = { pk2(acc[t][0], acc[t][1]), pk2(acc[t][2], acc[t][3]),
                  pk2(acc[t][4], acc[t][5]), pk2(acc[t][6], acc[t][7]) };
    *(uint4*)(ob + (size_t)o * N_) = pkd;
  }
}

// ---------------- flash attention: one 16-query tile per wave ----------------
// S^T = K^T x Q  (cols = queries -> lane-local softmax state), O^T = V x P^T.
// P^T in D-layout maps to the B-operand with one half-register xor-16 swap.
__global__ void attn_kernel(const unsigned short* __restrict__ qT,
                            const unsigned short* __restrict__ kT,
                            const unsigned short* __restrict__ vv,
                            unsigned short* __restrict__ ao){
  int wid = blockIdx.x * (blockDim.x >> 5) + (threadIdx.x >> 5);
  int b = wid >> 7;
  int rem = wid & 127;
  int h = rem >> 6;
  int i0 = (rem & 63) << 4;
  const float scale = 0.04419417382415922f;  // (C*HEADS)^-0.5
  int lane = threadIdx.x & 31;
  bool lo = (lane & 16) == 0;

  const unsigned short* qb_base = qT + (size_t)b * N_ * C_ + (size_t)i0 * C_ + h * HD;
  v16bf qb0 = load_B(qb_base +  0, C_);
  v16bf qb1 = load_B(qb_base + 32, C_);
  v16bf qb2 = load_B(qb_base + 64, C_);
  v16bf qb3 = load_B(qb_base + 96, C_);

  const unsigned short* kb_base = kT + (size_t)b * N_ * C_ + h * HD;
  const unsigned short* vb_base = vv + (size_t)b * C_ * N_ + (size_t)(h * HD) * N_;

  v8f oacc[8] = {};
  float m = -INFINITY, l = 0.f;

  for (int j0 = 0; j0 < N_; j0 += 32){
    const unsigned short* ka = kb_base + (size_t)j0 * C_;
    v8f s0 = {}, s1 = {};
    s0 = wmma_bf16(load_A(ka +  0, C_), qb0, s0);
    s1 = wmma_bf16(load_A(ka + 16 * C_ +  0, C_), qb0, s1);
    s0 = wmma_bf16(load_A(ka + 32, C_), qb1, s0);
    s1 = wmma_bf16(load_A(ka + 16 * C_ + 32, C_), qb1, s1);
    s0 = wmma_bf16(load_A(ka + 64, C_), qb2, s0);
    s1 = wmma_bf16(load_A(ka + 16 * C_ + 64, C_), qb2, s1);
    s0 = wmma_bf16(load_A(ka + 96, C_), qb3, s0);
    s1 = wmma_bf16(load_A(ka + 16 * C_ + 96, C_), qb3, s1);

    // online softmax over this 32-key chunk (per lane: 16 keys of its query column)
    float cm = -INFINITY;
    #pragma unroll
    for (int r = 0; r < 8; ++r){
      s0[r] *= scale; s1[r] *= scale;
      cm = fmaxf(cm, fmaxf(s0[r], s1[r]));
    }
    cm = fmaxf(cm, __shfl_xor(cm, 16, 32));
    float nm = fmaxf(m, cm);
    float corr = __expf(m - nm);
    float p0[8], p1[8], cs = 0.f;
    #pragma unroll
    for (int r = 0; r < 8; ++r){
      p0[r] = __expf(s0[r] - nm);
      p1[r] = __expf(s1[r] - nm);
      cs += p0[r] + p1[r];
    }
    cs += __shfl_xor(cs, 16, 32);
    l = l * corr + cs;
    m = nm;
    #pragma unroll
    for (int t = 0; t < 8; ++t)
      #pragma unroll
      for (int r = 0; r < 8; ++r) oacc[t][r] *= corr;

    // pack P^T into B-operand layout: swap upper halves between lane pairs (i, i+16)
    ABreg pt;
    #pragma unroll
    for (int d = 0; d < 4; ++d){
      unsigned a  = pk2(p0[2 * d], p0[2 * d + 1]);   // tile0 rows (j within 0..15)
      unsigned bb = pk2(p1[2 * d], p1[2 * d + 1]);   // tile1 rows (j within 16..31)
      unsigned send = lo ? bb : a;
      unsigned recv = __shfl_xor(send, 16, 32);
      pt.u[d]     = lo ? a : recv;
      pt.u[4 + d] = lo ? recv : bb;
    }

    #pragma unroll
    for (int t = 0; t < 8; ++t)
      oacc[t] = wmma_bf16(load_A(vb_base + (size_t)(t * 16) * N_ + j0, N_), pt.v, oacc[t]);
  }

  float invl = 1.0f / l;
  unsigned short* ob = ao + (size_t)b * N_ * C_ + (size_t)(i0 + (lane & 15)) * C_ + ((lane >> 4) << 3);
  #pragma unroll
  for (int t = 0; t < 8; ++t){
    uint4 pkd = { pk2(oacc[t][0] * invl, oacc[t][1] * invl),
                  pk2(oacc[t][2] * invl, oacc[t][3] * invl),
                  pk2(oacc[t][4] * invl, oacc[t][5] * invl),
                  pk2(oacc[t][6] * invl, oacc[t][7] * invl) };
    *(uint4*)(ob + t * 16) = pkd;
  }
}

// ---------------- out = Wp @ attn_out + residual (fp32 store, [b][o][n]) ----------------
__global__ void proj_final_kernel(const unsigned short* __restrict__ wpbf,
                                  const unsigned short* __restrict__ ao,
                                  const float* __restrict__ x,
                                  float* __restrict__ out){
  int wid = blockIdx.x * (blockDim.x >> 5) + (threadIdx.x >> 5);
  int b = wid >> 8;
  int rem = wid & 255;
  int o0 = (rem >> 4) << 4;
  int n0 = (rem & 15) << 6;
  const unsigned short* ab = ao + (size_t)b * N_ * C_;
  v8f acc[4] = {};
  for (int cc = 0; cc < C_; cc += 32){
    v16bf a = load_A(wpbf + (size_t)o0 * C_ + cc, C_);
    #pragma unroll
    for (int t = 0; t < 4; ++t)
      acc[t] = wmma_bf16(a, load_B(ab + (size_t)(n0 + t * 16) * C_ + cc, C_), acc[t]);
  }
  int lane = threadIdx.x & 31;
  int ob0 = o0 + ((lane >> 4) << 3);
  const float* xr = x + (size_t)b * C_ * N_;
  float* orow = out + (size_t)b * C_ * N_;
  #pragma unroll
  for (int t = 0; t < 4; ++t){
    int n = n0 + t * 16 + (lane & 15);
    #pragma unroll
    for (int r = 0; r < 8; ++r){
      size_t idx = (size_t)(ob0 + r) * N_ + n;
      orow[idx] = acc[t][r] + xr[idx];
    }
  }
}

// ---------------- host launch ----------------
extern "C" void kernel_launch(void* const* d_in, const int* in_sizes, int n_in,
                              void* d_out, int out_size, void* d_ws, size_t ws_size,
                              hipStream_t stream){
  const float* x   = (const float*)d_in[0];
  const float* gnw = (const float*)d_in[1];
  const float* gnb = (const float*)d_in[2];
  const float* wq  = (const float*)d_in[3];
  const float* wk  = (const float*)d_in[4];
  const float* wv  = (const float*)d_in[5];
  const float* wp  = (const float*)d_in[6];
  float* out = (float*)d_out;

  char* ws = (char*)d_ws;
  size_t off = 0;
  auto wsalloc = [&](size_t bytes) -> void* {
    void* p = ws + off;
    off = (off + bytes + 255) & ~(size_t)255;
    return p;
  };
  const size_t MAT = (size_t)B_ * N_ * C_ * sizeof(unsigned short);   // 8 MB
  const size_t WMAT = (size_t)C_ * C_ * sizeof(unsigned short);       // 128 KB
  unsigned short* xnT  = (unsigned short*)wsalloc(MAT);  // also reused as `ao` after projections
  unsigned short* qT   = (unsigned short*)wsalloc(MAT);
  unsigned short* kT   = (unsigned short*)wsalloc(MAT);
  unsigned short* vbuf = (unsigned short*)wsalloc(MAT);
  unsigned short* wqb  = (unsigned short*)wsalloc(WMAT);
  unsigned short* wkb  = (unsigned short*)wsalloc(WMAT);
  unsigned short* wvb  = (unsigned short*)wsalloc(WMAT);
  unsigned short* wpb  = (unsigned short*)wsalloc(WMAT);
  unsigned short* ao = xnT;   // xnT dead after q/k/v projections; stream order makes reuse safe

  gn_kernel<<<B_ * GROUPS, 256, 0, stream>>>(x, gnw, gnb, xnT);
  cvt_kernel<<<(C_ * C_) / 256, 256, 0, stream>>>(wq, wqb, C_ * C_);
  cvt_kernel<<<(C_ * C_) / 256, 256, 0, stream>>>(wk, wkb, C_ * C_);
  cvt_kernel<<<(C_ * C_) / 256, 256, 0, stream>>>(wv, wvb, C_ * C_);
  cvt_kernel<<<(C_ * C_) / 256, 256, 0, stream>>>(wp, wpb, C_ * C_);

  // 4096 waves each: 16o x 64n (or 16n x 64o) tiles, 4 waves / block
  proj_qk_kernel<<<1024, 128, 0, stream>>>(wqb, xnT, qT);
  proj_qk_kernel<<<1024, 128, 0, stream>>>(wkb, xnT, kT);
  proj_v_kernel <<<1024, 128, 0, stream>>>(xnT, wvb, vbuf);

  // 2048 waves: one (b, h, 16-query tile) each
  attn_kernel<<<512, 128, 0, stream>>>(qT, kT, vbuf, ao);

  proj_final_kernel<<<1024, 128, 0, stream>>>(wpb, ao, x, out);
}
